// MultiHeadedSpatialAttention_15633680957731
// MI455X (gfx1250) — matrix operand order repaired and checked by reading, not compiled, not run on hardware
//
#include <hip/hip_runtime.h>

// ---------------- problem constants (from reference) ----------------
#define B_    4
#define C_    64
#define H_    128
#define W_    128
#define HW_   (H_ * W_)
#define HEADS_ 4
#define MID_  16                      // channels per head
#define SECSTRIDE_ (B_ * C_ * HW_)    // floats per Q/K/V section

// ---------------- WMMA types ----------------
typedef __bf16 bf16_t;
typedef bf16_t v16bf __attribute__((ext_vector_type(16)));
typedef float  v8f   __attribute__((ext_vector_type(8)));

union BFrag { v16bf v; unsigned int u[8]; };

__device__ __forceinline__ unsigned short f2bf(float f) {
    unsigned u = __float_as_uint(f);
    return (unsigned short)((u + 0x7FFFu + ((u >> 16) & 1u)) >> 16); // RNE
}
__device__ __forceinline__ unsigned pack_bf(float lo, float hi) {
    return (unsigned)f2bf(lo) | ((unsigned)f2bf(hi) << 16);
}

// -------- gfx1250 async global->LDS path (guarded, safe fallback) --------
#if defined(__has_builtin)
#  if __has_builtin(__builtin_amdgcn_global_load_async_to_lds_b32)
#    define HAVE_ASYNC_LDS 1
#  endif
#endif
#ifndef HAVE_ASYNC_LDS
#  define HAVE_ASYNC_LDS 0
#endif

#if HAVE_ASYNC_LDS
typedef __attribute__((address_space(1))) int g_i32;   // global (AS1) int*
typedef __attribute__((address_space(3))) int l_i32;   // LDS    (AS3) int*
__device__ __forceinline__ void async_wait0() {
#  if __has_builtin(__builtin_amdgcn_s_wait_asynccnt)
    __builtin_amdgcn_s_wait_asynccnt(0);
#  else
    asm volatile("s_wait_asynccnt 0" ::: "memory");
#  endif
}
#endif

// ---------------- stage 0: weight/bias prep (fp32 -> bf16, concat) ----------------
__global__ __launch_bounds__(256) void prep_kernel(
    const float* __restrict__ Wq, const float* __restrict__ bq,
    const float* __restrict__ Wk, const float* __restrict__ bk,
    const float* __restrict__ Wv, const float* __restrict__ bv,
    unsigned short* __restrict__ wbf, float* __restrict__ bias)
{
    int i = blockIdx.x * blockDim.x + threadIdx.x;
    if (i < 192 * 64) {
        int m = i >> 6, c = i & 63;
        const float* src = (m < 64) ? Wq : (m < 128 ? Wk : Wv);
        wbf[i] = f2bf(src[(m & 63) * 64 + c]);
    }
    if (i < 192) {
        bias[i] = (i < 64) ? bq[i] : (i < 128 ? bk[i - 64] : bv[i - 128]);
    }
}

// ---------------- stage 1: fused QKV 1x1-conv GEMM via bf16 WMMA ----------------
// D[m,n] = sum_k W[m,k] * x[k,n] + bias[m],  m in [0,192) = {Q,K,V}x64, n = pixel
__global__ __launch_bounds__(256) void qkv_gemm_wmma(
    const float* __restrict__ x, const unsigned short* __restrict__ wbf,
    const float* __restrict__ bias, float* __restrict__ qkv)
{
    const int blk  = blockIdx.x;
    const int b    = blk >> 7;          // 128 tiles per batch
    const int tile = blk & 127;
    const int wave = threadIdx.x >> 5;
    const int lane = threadIdx.x & 31;
    const int n0   = tile * 128 + wave * 16;
    const int nl   = lane & 15;         // column within N-tile / M within A-tile
    const int kh   = lane >> 4;         // K-half selector (contiguous order k = 16*kh + s)

    // ---- B fragments (x tile, 64 x 16 pixels, split K into 0..31 / 32..63) ----
    BFrag bLo, bHi;
    const float* xb = x + (size_t)b * C_ * HW_ + n0 + nl;
#pragma unroll
    for (int v = 0; v < 8; ++v) {
        const int k0 = kh * 16 + 2 * v;
        bLo.u[v] = pack_bf(xb[(size_t)(k0)      * HW_], xb[(size_t)(k0 + 1)  * HW_]);
        bHi.u[v] = pack_bf(xb[(size_t)(k0 + 32) * HW_], xb[(size_t)(k0 + 33) * HW_]);
    }

#pragma unroll
    for (int mt = 0; mt < 12; ++mt) {
        BFrag aLo, aHi;
        const int m = mt * 16 + nl;
#pragma unroll
        for (int v = 0; v < 8; ++v) {
            const int k = (kh << 4) + (v << 1);
            aLo.u[v] = *(const unsigned int*)(wbf + m * 64 + k);
            aHi.u[v] = *(const unsigned int*)(wbf + m * 64 + k + 32);
        }

        v8f acc = {};
        acc = __builtin_amdgcn_wmma_f32_16x16x32_bf16(false, aLo.v, false, bLo.v,
                                                      (short)0, acc, false, false);
        acc = __builtin_amdgcn_wmma_f32_16x16x32_bf16(false, aHi.v, false, bHi.v,
                                                      (short)0, acc, false, false);

        const int sec = mt >> 2;                 // 0=Q, 1=K, 2=V
        float* dst = qkv + (size_t)sec * SECSTRIDE_;
#pragma unroll
        for (int r = 0; r < 8; ++r) {
            const int mr = mt * 16 + r + kh * 8;           // row in 192
            const int ch = (mt & 3) * 16 + r + kh * 8;     // channel in 64
            dst[(size_t)(b * C_ + ch) * HW_ + n0 + nl] = acc[r] + bias[mr];
        }
    }
}

// ---------------- stage 2: 7x7 local attention + softmax + residual ----------------
#define TS_   16
#define HALO_ 3
#define TSH_  22          // TS + 2*HALO
#define CPAD_ 20          // per-pixel channel stride: conflict-free b128 LDS loads
#define TILE_ELEMS_ (MID_ * TSH_ * TSH_)

// Fill the [y][x][c] LDS tile from one 16-channel plane set (zero pad at borders).
// Optionally prefetch the matching elements of a second plane set (V during K fill).
__device__ __forceinline__ void fill_tile(float* tile_kv,
                                          const float* __restrict__ plane,
                                          const float* __restrict__ pf,
                                          size_t planeBase, int ty0, int tx0, int tid)
{
    for (int i = tid; i < TILE_ELEMS_; i += 256) {
        const int c   = i / (TSH_ * TSH_);
        const int rem = i - c * (TSH_ * TSH_);
        const int yy  = rem / TSH_;
        const int xx  = rem - yy * TSH_;
        const int gy  = ty0 + yy - HALO_;
        const int gx  = tx0 + xx - HALO_;
        const int off = (yy * TSH_ + xx) * CPAD_ + c;
        if (gy >= 0 && gy < H_ && gx >= 0 && gx < W_) {
            const size_t idx = planeBase + (size_t)c * HW_ + (size_t)gy * W_ + gx;
#if HAVE_ASYNC_LDS
            __builtin_amdgcn_global_load_async_to_lds_b32(
                (g_i32*)(plane + idx), (l_i32*)&tile_kv[off], 0, 0);
#else
            tile_kv[off] = plane[idx];
#endif
            if (pf) __builtin_prefetch(pf + idx, 0, 0);
        } else {
            tile_kv[off] = 0.f;
        }
    }
#if HAVE_ASYNC_LDS
    async_wait0();                     // drain ASYNCcnt before the barrier
#endif
}

__global__ __launch_bounds__(256) void attn_kernel(
    const float* __restrict__ x,  const float* __restrict__ qp,
    const float* __restrict__ kp, const float* __restrict__ vp,
    float* __restrict__ out)
{
    __shared__ __align__(16) float tile_kv[TSH_ * TSH_ * CPAD_];   // 38.7 KB, K then V

    const int blk  = blockIdx.x;
    const int tile = blk & 63;                 // 8x8 tiles of 16x16
    const int g    = (blk >> 6) & 3;           // head
    const int b    = blk >> 8;                 // batch
    const int ty0  = (tile >> 3) * TS_;
    const int tx0  = (tile & 7) * TS_;
    const int tid  = threadIdx.x;

    const size_t planeBase = (size_t)(b * C_ + g * MID_) * HW_;

    const int py = tid >> 4, px = tid & 15;
    const size_t pix = (size_t)(ty0 + py) * W_ + (tx0 + px);
    __builtin_prefetch(x + planeBase + pix, 0, 1);   // residual read -> global_prefetch

    // ---- phase 1: K tile in LDS (prefetch V in its shadow) ----
    fill_tile(tile_kv, kp, vp, planeBase, ty0, tx0, tid);
    __syncthreads();

    // ---- per-pixel Q in registers ----
    float q[MID_];
#pragma unroll
    for (int c = 0; c < MID_; ++c) q[c] = qp[planeBase + (size_t)c * HW_ + pix];

    // ---- logits: A[p] = Q . K_p, 4x ds_load_b128 per tap ----
    float a[49];
#pragma unroll
    for (int dy = 0; dy < 7; ++dy)
#pragma unroll
        for (int dx = 0; dx < 7; ++dx) {
            const float4* kr =
                (const float4*)&tile_kv[((py + dy) * TSH_ + (px + dx)) * CPAD_];
            const float4 k0 = kr[0], k1 = kr[1], k2 = kr[2], k3 = kr[3];
            float s = 0.f;
            s = fmaf(q[0],  k0.x, s); s = fmaf(q[1],  k0.y, s);
            s = fmaf(q[2],  k0.z, s); s = fmaf(q[3],  k0.w, s);
            s = fmaf(q[4],  k1.x, s); s = fmaf(q[5],  k1.y, s);
            s = fmaf(q[6],  k1.z, s); s = fmaf(q[7],  k1.w, s);
            s = fmaf(q[8],  k2.x, s); s = fmaf(q[9],  k2.y, s);
            s = fmaf(q[10], k2.z, s); s = fmaf(q[11], k2.w, s);
            s = fmaf(q[12], k3.x, s); s = fmaf(q[13], k3.y, s);
            s = fmaf(q[14], k3.z, s); s = fmaf(q[15], k3.w, s);
            a[dy * 7 + dx] = s;
        }

    // ---- softmax over the 49 offsets ----
    float mx = a[0];
#pragma unroll
    for (int p = 1; p < 49; ++p) mx = fmaxf(mx, a[p]);
    float den = 0.f;
#pragma unroll
    for (int p = 0; p < 49; ++p) { a[p] = __expf(a[p] - mx); den += a[p]; }
    const float inv = __builtin_amdgcn_rcpf(den);

    // ---- phase 2: reuse the same LDS buffer for the V tile ----
    __syncthreads();                                  // all K reads done
    fill_tile(tile_kv, vp, (const float*)0, planeBase, ty0, tx0, tid);
    __syncthreads();

    // ---- y = sum_p A[p] * V_p ----
    float y[MID_];
#pragma unroll
    for (int c = 0; c < MID_; ++c) y[c] = 0.f;
#pragma unroll
    for (int dy = 0; dy < 7; ++dy)
#pragma unroll
        for (int dx = 0; dx < 7; ++dx) {
            const float w = a[dy * 7 + dx];
            const float4* vr =
                (const float4*)&tile_kv[((py + dy) * TSH_ + (px + dx)) * CPAD_];
            const float4 v0 = vr[0], v1 = vr[1], v2 = vr[2], v3 = vr[3];
            y[0]  = fmaf(w, v0.x, y[0]);  y[1]  = fmaf(w, v0.y, y[1]);
            y[2]  = fmaf(w, v0.z, y[2]);  y[3]  = fmaf(w, v0.w, y[3]);
            y[4]  = fmaf(w, v1.x, y[4]);  y[5]  = fmaf(w, v1.y, y[5]);
            y[6]  = fmaf(w, v1.z, y[6]);  y[7]  = fmaf(w, v1.w, y[7]);
            y[8]  = fmaf(w, v2.x, y[8]);  y[9]  = fmaf(w, v2.y, y[9]);
            y[10] = fmaf(w, v2.z, y[10]); y[11] = fmaf(w, v2.w, y[11]);
            y[12] = fmaf(w, v3.x, y[12]); y[13] = fmaf(w, v3.y, y[13]);
            y[14] = fmaf(w, v3.z, y[14]); y[15] = fmaf(w, v3.w, y[15]);
        }

    // ---- residual add + store ----
#pragma unroll
    for (int c = 0; c < MID_; ++c) {
        const size_t idx = planeBase + (size_t)c * HW_ + pix;
        out[idx] = x[idx] + y[c] * inv;
    }
}

// ---------------- host side ----------------
extern "C" void kernel_launch(void* const* d_in, const int* in_sizes, int n_in,
                              void* d_out, int out_size, void* d_ws, size_t ws_size,
                              hipStream_t stream)
{
    const float* x  = (const float*)d_in[0];
    const float* Wq = (const float*)d_in[1];
    const float* bq = (const float*)d_in[2];
    const float* Wk = (const float*)d_in[3];
    const float* bk = (const float*)d_in[4];
    const float* Wv = (const float*)d_in[5];
    const float* bv = (const float*)d_in[6];

    char* ws = (char*)d_ws;
    unsigned short* wbf = (unsigned short*)ws;            // 192*64*2 = 24576 B
    float* bias = (float*)(ws + 24576);                   // 192*4 B
    float* qkv  = (float*)(ws + 32768);                   // 3 * SECSTRIDE_ floats (~50 MB, L2-resident)
    float* qp   = qkv;
    float* kpp  = qkv + (size_t)SECSTRIDE_;
    float* vpp  = qkv + (size_t)2 * SECSTRIDE_;

    prep_kernel<<<48, 256, 0, stream>>>(Wq, bq, Wk, bk, Wv, bv, wbf, bias);
    qkv_gemm_wmma<<<512, 256, 0, stream>>>(x, wbf, bias, qkv);
    attn_kernel<<<B_ * HEADS_ * 64, 256, 0, stream>>>(x, qp, kpp, vpp, (float*)d_out);
}
